// EncoderBlock_83571473645734
// MI455X (gfx1250) — matrix operand order repaired and checked by reading, not compiled, not run on hardware
//
#include <hip/hip_runtime.h>
#include <stdint.h>

// ---------------------------------------------------------------------------
// CDNA5 (gfx1250) encoder block: bf16 WMMA everywhere, flash attention.
// wave32; WMMA shape 16x16x32 bf16 -> f32 accum.
// ---------------------------------------------------------------------------

typedef __attribute__((ext_vector_type(16))) __bf16 v16bf;
typedef __attribute__((ext_vector_type(8)))  float  v8f;

#define D_MODEL 1024
#define SEQ     4096
#define NH      8
#define HD      128
#define DFF     4096
#define LN_EPS  1e-5f
#define QSCALE  0.08838834764831845f  // 1/sqrt(128)

struct Frag { union { v16bf v; uint4 q[2]; unsigned u[8]; }; };

__device__ __forceinline__ unsigned short f2bf(float f) {
  union { float f; unsigned u; } x; x.f = f;
  unsigned r = 0x7FFFu + ((x.u >> 16) & 1u);   // round-to-nearest-even
  return (unsigned short)((x.u + r) >> 16);
}
__device__ __forceinline__ unsigned pack2(float a, float b) {
  return (unsigned)f2bf(a) | ((unsigned)f2bf(b) << 16);
}
__device__ __forceinline__ v8f wmma_bf16(const v16bf& a, const v16bf& b, const v8f& c) {
  return __builtin_amdgcn_wmma_f32_16x16x32_bf16(false, a, false, b, (short)0, c, false, false);
}

// wave32 half-swap (lane i <-> lane i^16) as pure VALU: v_permlanex16_b32
// identity nibble selectors -> lane i reads lane (i^16)
__device__ __forceinline__ int xor16_i32(int v) {
  return __builtin_amdgcn_permlanex16(v, v, 0x76543210, 0xfedcba98, false, false);
}
__device__ __forceinline__ float xor16_f32(float v) {
  return __int_as_float(xor16_i32(__float_as_int(v)));
}

// ---------------- weight fp32 [K][N] -> bf16 transposed [N][K] -------------
__global__ void k_w2bf_t(const float* __restrict__ w, unsigned short* __restrict__ wt,
                         int K, int N) {
  int i = blockIdx.x * 256 + threadIdx.x;
  if (i >= K * N) return;
  int n = i / K, k = i - n * K;
  wt[i] = f2bf(w[k * N + n]);
}

// ---------------- LayerNorm (optional residual), fp32 + bf16 outputs -------
__global__ void k_layernorm(const float* __restrict__ x, const float* __restrict__ res,
                            const float* __restrict__ g, const float* __restrict__ b,
                            float* __restrict__ yf, unsigned short* __restrict__ ybf) {
  const int row = blockIdx.x, tid = threadIdx.x;   // 256 threads, 4 elems each
  __shared__ float sh[256];
  float4 v = ((const float4*)(x + (size_t)row * D_MODEL))[tid];
  if (res) {
    float4 r = ((const float4*)(res + (size_t)row * D_MODEL))[tid];
    v.x += r.x; v.y += r.y; v.z += r.z; v.w += r.w;
  }
  sh[tid] = v.x + v.y + v.z + v.w;
  __syncthreads();
  for (int o = 128; o > 0; o >>= 1) { if (tid < o) sh[tid] += sh[tid + o]; __syncthreads(); }
  const float mu = sh[0] * (1.0f / D_MODEL);
  __syncthreads();
  float dx = v.x - mu, dy = v.y - mu, dz = v.z - mu, dw = v.w - mu;
  sh[tid] = dx * dx + dy * dy + dz * dz + dw * dw;
  __syncthreads();
  for (int o = 128; o > 0; o >>= 1) { if (tid < o) sh[tid] += sh[tid + o]; __syncthreads(); }
  const float rstd = rsqrtf(sh[0] * (1.0f / D_MODEL) + LN_EPS);
  float4 gg = ((const float4*)g)[tid], bb = ((const float4*)b)[tid];
  float o0 = dx * rstd * gg.x + bb.x;
  float o1 = dy * rstd * gg.y + bb.y;
  float o2 = dz * rstd * gg.z + bb.z;
  float o3 = dw * rstd * gg.w + bb.w;
  if (yf) ((float4*)(yf + (size_t)row * D_MODEL))[tid] = make_float4(o0, o1, o2, o3);
  uint2 p; p.x = pack2(o0, o1); p.y = pack2(o2, o3);
  ((uint2*)(ybf + (size_t)row * D_MODEL))[tid] = p;
}

// ---------------- WMMA GEMM: C[M,N] = A(bf16,[M][K]) * Bt(bf16,[N][K]) -----
// 4 waves/block, each wave computes a 16(M)x64(N) strip.
// MODE 0: QKV epilogue (split q*scale, k, v^T as bf16)
// MODE 1: exact GELU -> bf16      MODE 2: fp32 out
template <int MODE>
__global__ __launch_bounds__(128, 1) void k_gemm_wmma(
    const unsigned short* __restrict__ A,
    const unsigned short* __restrict__ Bt,
    const float* __restrict__ bias,
    unsigned short* __restrict__ o0,
    unsigned short* __restrict__ o1,
    unsigned short* __restrict__ o2,
    float* __restrict__ of,
    int M, int N, int K) {
  const int lane = threadIdx.x & 31, wave = threadIdx.x >> 5;
  const int lq = lane & 15, hi = lane >> 4;
  const int m0 = blockIdx.y * 64 + wave * 16;
  const int n0 = blockIdx.x * 64;
  v8f acc[4] = {};
  const unsigned short* pa_base = A + (size_t)(m0 + lq) * K;
  for (int k0 = 0; k0 < K; k0 += 32) {
    Frag a;
    const unsigned short* pa = pa_base + k0 + hi * 8;
    a.q[0] = *(const uint4*)pa;
    a.q[1] = *(const uint4*)(pa + 16);
    if (k0 + 32 < K) __builtin_prefetch(pa + 32, 0, 1);   // global_prefetch_b8
#pragma unroll
    for (int t = 0; t < 4; ++t) {
      Frag bf;
      const unsigned short* pb = Bt + (size_t)(n0 + 16 * t + lq) * K + k0 + hi * 8;
      bf.q[0] = *(const uint4*)pb;
      bf.q[1] = *(const uint4*)(pb + 16);
      if (k0 + 32 < K) __builtin_prefetch(pb + 32, 0, 1);
      acc[t] = wmma_bf16(a.v, bf.v, acc[t]);
    }
  }
#pragma unroll
  for (int t = 0; t < 4; ++t) {
    const int col = n0 + 16 * t + lq;
    const float bia = bias[col];
#pragma unroll
    for (int e = 0; e < 8; ++e) {
      const int row = m0 + e + hi * 8;
      float val = acc[t][e] + bia;
      if (MODE == 0) {
        const int sel = col >> 10, head = (col & 1023) >> 7, dim = col & 127;
        if (sel == 0)      o0[((size_t)head * SEQ + row) * HD + dim] = f2bf(val * QSCALE);
        else if (sel == 1) o1[((size_t)head * SEQ + row) * HD + dim] = f2bf(val);
        else               o2[((size_t)head * HD + dim) * SEQ + row] = f2bf(val);  // V^T
      } else if (MODE == 1) {
        float gv = 0.5f * val * (1.0f + erff(val * 0.70710678118654752f));
        o0[(size_t)row * N + col] = f2bf(gv);
      } else {
        of[(size_t)row * N + col] = val;
      }
    }
  }
}

// ---------------- Flash attention: 1 wave per (head, 16-query tile) --------
// S^T = K * Q^T  (row-reduce = intra-lane + permlanex16)
// O^T = V^T * P^T (P^T B-frag built per-lane from S^T C-layout, 8 permlanes)
// launch_bounds(32,1): single wave per block -> allow full VGPR budget, no spills
__global__ __launch_bounds__(32, 1) void k_flash_attn(
    const unsigned short* __restrict__ qb,
    const unsigned short* __restrict__ kb,
    const unsigned short* __restrict__ vt,
    float* __restrict__ attn) {
  const int head = blockIdx.y;
  const int q0 = blockIdx.x * 16;
  const int lane = threadIdx.x & 31;
  const int lq = lane & 15, hi = lane >> 4;

  Frag bq[4];  // Q^T B-fragments, 4 dim-chunks of 32
#pragma unroll
  for (int c = 0; c < 4; ++c) {
    const unsigned short* p = qb + ((size_t)head * SEQ + q0 + lq) * HD + 32 * c + hi * 8;
    bq[c].q[0] = *(const uint4*)p;
    bq[c].q[1] = *(const uint4*)(p + 16);
  }
  v8f ot[8] = {};                 // O^T accum: 8 dim-chunks of 16 (C-layout: col=query=lq)
  float mrun = -1e30f, ssum = 0.0f;
  const unsigned short* kbase = kb + (size_t)head * SEQ * HD;
  const unsigned short* vbase = vt + (size_t)head * HD * SEQ;

  for (int kb0 = 0; kb0 < SEQ; kb0 += 32) {
    v8f st0 = {}, st1 = {};       // S^T tiles: keys kb0..+15 and kb0+16..+31
#pragma unroll
    for (int c = 0; c < 4; ++c) {
      Frag a0, a1;
      const unsigned short* p0 = kbase + (size_t)(kb0 + lq) * HD + 32 * c + hi * 8;
      a0.q[0] = *(const uint4*)p0; a0.q[1] = *(const uint4*)(p0 + 16);
      const unsigned short* p1 = kbase + (size_t)(kb0 + 16 + lq) * HD + 32 * c + hi * 8;
      a1.q[0] = *(const uint4*)p1; a1.q[1] = *(const uint4*)(p1 + 16);
      st0 = wmma_bf16(a0.v, bq[c].v, st0);
      st1 = wmma_bf16(a1.v, bq[c].v, st1);
    }
    // online softmax over 32 keys for query lq (both half-waves converge)
    float lmax = -1e30f;
#pragma unroll
    for (int e = 0; e < 8; ++e) lmax = fmaxf(lmax, fmaxf(st0[e], st1[e]));
    lmax = fmaxf(lmax, xor16_f32(lmax));
    const float mnew = fmaxf(mrun, lmax);
    const float alpha = __expf(mrun - mnew);
    float p0f[8], p1f[8], rs = 0.0f;
#pragma unroll
    for (int e = 0; e < 8; ++e) {
      p0f[e] = __expf(st0[e] - mnew);
      p1f[e] = __expf(st1[e] - mnew);
      rs += p0f[e] + p1f[e];
    }
    rs += xor16_f32(rs);
    ssum = ssum * alpha + rs;
    mrun = mnew;
#pragma unroll
    for (int t = 0; t < 8; ++t) ot[t] = ot[t] * alpha;
    // build P^T (32k x 16q) B-fragment: per-lane packs + one half-wave swap
    unsigned pk0[4], pk1[4];
#pragma unroll
    for (int i = 0; i < 4; ++i) {
      pk0[i] = pack2(p0f[2 * i], p0f[2 * i + 1]);
      pk1[i] = pack2(p1f[2 * i], p1f[2 * i + 1]);
    }
    Frag bp;
#pragma unroll
    for (int i = 0; i < 4; ++i) {
      unsigned e0 = (unsigned)xor16_i32((int)pk0[i]);
      unsigned e1 = (unsigned)xor16_i32((int)pk1[i]);
      bp.u[i]     = hi ? e1 : pk0[i];
      bp.u[i + 4] = hi ? pk1[i] : e0;
    }
    // O^T += V^T(16d x 32k) * P^T : V stored transposed -> contiguous A loads
#pragma unroll
    for (int t = 0; t < 8; ++t) {
      Frag av;
      const unsigned short* pv = vbase + (size_t)(16 * t + lq) * SEQ + kb0 + hi * 16;
      av.q[0] = *(const uint4*)pv; av.q[1] = *(const uint4*)(pv + 8);
      ot[t] = wmma_bf16(av.v, bp.v, ot[t]);
    }
  }
  const float inv = 1.0f / ssum;
  float* outrow = attn + (size_t)(q0 + lq) * D_MODEL + head * HD;
#pragma unroll
  for (int t = 0; t < 8; ++t)
#pragma unroll
    for (int e = 0; e < 8; ++e)
      outrow[16 * t + hi * 8 + e] = ot[t][e] * inv;
}

// ---------------------------------------------------------------------------
extern "C" void kernel_launch(void* const* d_in, const int* in_sizes, int n_in,
                              void* d_out, int out_size, void* d_ws, size_t ws_size,
                              hipStream_t stream) {
  const float* x     = (const float*)d_in[0];
  const float* ln1_g = (const float*)d_in[1];
  const float* ln1_b = (const float*)d_in[2];
  const float* qkv_w = (const float*)d_in[3];
  const float* qkv_b = (const float*)d_in[4];
  const float* ln2_g = (const float*)d_in[5];
  const float* ln2_b = (const float*)d_in[6];
  const float* w1    = (const float*)d_in[7];
  const float* b1    = (const float*)d_in[8];
  const float* w2    = (const float*)d_in[9];
  const float* b2    = (const float*)d_in[10];
  float* out = (float*)d_out;

  char* ws = (char*)d_ws;
  size_t off = 0;
  auto alloc = [&](size_t bytes) -> void* {
    void* p = ws + off; off += (bytes + 255) & ~(size_t)255; return p;
  };
  float*          h1f  = (float*)alloc((size_t)SEQ * D_MODEL * 4);          // LN1 out fp32
  unsigned short* h1b  = (unsigned short*)alloc((size_t)SEQ * D_MODEL * 2); // LN1 out bf16
  unsigned short* qwt  = (unsigned short*)alloc((size_t)D_MODEL * 3 * D_MODEL * 2);
  unsigned short* w1t  = (unsigned short*)alloc((size_t)D_MODEL * DFF * 2);
  unsigned short* w2t  = (unsigned short*)alloc((size_t)DFF * D_MODEL * 2);
  unsigned short* qbuf = (unsigned short*)alloc((size_t)SEQ * D_MODEL * 2); // [H][S][hd]
  unsigned short* kbuf = (unsigned short*)alloc((size_t)SEQ * D_MODEL * 2); // [H][S][hd]
  unsigned short* vtb  = (unsigned short*)alloc((size_t)SEQ * D_MODEL * 2); // [H][hd][S]
  float*          attn = (float*)alloc((size_t)SEQ * D_MODEL * 4);
  unsigned short* h2b  = (unsigned short*)alloc((size_t)SEQ * D_MODEL * 2);
  unsigned short* gbuf = (unsigned short*)alloc((size_t)SEQ * DFF * 2);
  (void)ws_size; // total ~126 MB

  // weights -> bf16, transposed to [N][K]
  k_w2bf_t<<<(D_MODEL * 3 * D_MODEL) / 256, 256, 0, stream>>>(qkv_w, qwt, D_MODEL, 3 * D_MODEL);
  k_w2bf_t<<<(D_MODEL * DFF) / 256, 256, 0, stream>>>(w1, w1t, D_MODEL, DFF);
  k_w2bf_t<<<(DFF * D_MODEL) / 256, 256, 0, stream>>>(w2, w2t, DFF, D_MODEL);

  // LN1
  k_layernorm<<<SEQ, 256, 0, stream>>>(x, nullptr, ln1_g, ln1_b, h1f, h1b);

  // QKV GEMM (M=4096, N=3072, K=1024) -> q*scale, k, v^T (bf16)
  k_gemm_wmma<0><<<dim3((3 * D_MODEL) / 64, SEQ / 64), 128, 0, stream>>>(
      h1b, qwt, qkv_b, qbuf, kbuf, vtb, nullptr, SEQ, 3 * D_MODEL, D_MODEL);

  // Flash attention -> attn [S][D] fp32
  k_flash_attn<<<dim3(SEQ / 16, NH), 32, 0, stream>>>(qbuf, kbuf, vtb, attn);

  // residual (attn + LN1 out) -> LN2 -> bf16
  k_layernorm<<<SEQ, 256, 0, stream>>>(attn, h1f, ln2_g, ln2_b, nullptr, h2b);

  // FFN1: gelu(h2 @ w1 + b1) -> bf16 (M=4096, N=4096, K=1024)
  k_gemm_wmma<1><<<dim3(DFF / 64, SEQ / 64), 128, 0, stream>>>(
      h2b, w1t, b1, gbuf, nullptr, nullptr, nullptr, SEQ, DFF, D_MODEL);

  // FFN2: g @ w2 + b2 -> fp32 out (M=4096, N=1024, K=4096)
  k_gemm_wmma<2><<<dim3(D_MODEL / 64, SEQ / 64), 128, 0, stream>>>(
      gbuf, w2t, b2, nullptr, nullptr, nullptr, out, SEQ, D_MODEL, DFF);
}
